// MultiHeadedAttention_57827439673522
// MI455X (gfx1250) — compile-verified, matrix operand order checked by reading, not run on hardware
//
#include <hip/hip_runtime.h>
#include <hip/hip_bf16.h>
#include <math.h>

typedef __bf16 bf16;
typedef __attribute__((ext_vector_type(16))) __bf16 v16bf;
typedef __attribute__((ext_vector_type(8)))  __bf16 v8bf;
typedef __attribute__((ext_vector_type(8)))  float  v8f;

#define H_  12
#define D_  768
#define DH_ 64
#define S_  4096

union FragU { v16bf v; v8bf h[2]; };

__device__ __forceinline__ int lane_id() { return (int)(threadIdx.x & 31u); }

// Load a 16x32 bf16 A-fragment (or 32x16 B-fragment via its column-major view)
// from row-major memory with leading dimension `ld`.
// Lane l: row/col = l&15 ; element i holds k = (i>>3)*16 + ((l>>4)*8) + (i&7)
// => two contiguous 16-byte chunks per lane at k-offsets kh and kh+16.
__device__ __forceinline__ v16bf load_frag(const bf16* __restrict__ base, int ld) {
  const int l  = lane_id();
  const int rc = l & 15;
  const int kh = (l >> 4) << 3;
  const bf16* p = base + (size_t)rc * ld + kh;
  FragU f;
  f.h[0] = *reinterpret_cast<const v8bf*>(p);
  f.h[1] = *reinterpret_cast<const v8bf*>(p + 16);
  return f.v;
}

__device__ __forceinline__ v8f wmma_bf16(v16bf a, v16bf b, v8f c) {
  return __builtin_amdgcn_wmma_f32_16x16x32_bf16(false, a, false, b, (short)0, c,
                                                 false, false);
}

// ---------------- conversion / transpose kernels ----------------

__global__ void cvt_f32_bf16(const float* __restrict__ src, bf16* __restrict__ dst, int n) {
  int i = blockIdx.x * blockDim.x + threadIdx.x;
  int stride = gridDim.x * blockDim.x;
  for (; i < n; i += stride) dst[i] = (bf16)src[i];
}

// src [H][D][DH] fp32 -> dst [H][DH][D] bf16
__global__ void cvt_wT(const float* __restrict__ src, bf16* __restrict__ dst) {
  const int n = H_ * D_ * DH_;
  int i = blockIdx.x * blockDim.x + threadIdx.x;
  int stride = gridDim.x * blockDim.x;
  for (; i < n; i += stride) {
    int h   = i / (D_ * DH_);
    int rem = i - h * (D_ * DH_);
    int d   = rem / DH_;
    int e   = rem - d * DH_;
    dst[((size_t)h * DH_ + e) * D_ + d] = (bf16)src[i];
  }
}

// src [K=768][N=768] fp32 -> dst [N][K] bf16
__global__ void cvt_woT(const float* __restrict__ src, bf16* __restrict__ dst) {
  const int n = D_ * D_;
  int i = blockIdx.x * blockDim.x + threadIdx.x;
  int stride = gridDim.x * blockDim.x;
  for (; i < n; i += stride) {
    int kk = i / D_;
    int nn = i - kk * D_;
    dst[(size_t)nn * D_ + kk] = (bf16)src[i];
  }
}

// ---------------- QKV projection ----------------
// grid: (S/32, H, 3), block: 32 (1 wave). Each wave: 32x64 tile, K=768,
// software-pipelined (double-buffered fragments).
__global__ void qkv_proj_kernel(const bf16* __restrict__ xb,
                                const bf16* __restrict__ wqT,
                                const bf16* __restrict__ wkT,
                                const bf16* __restrict__ wvT,
                                const float* __restrict__ bq,
                                const float* __restrict__ bk,
                                const float* __restrict__ bv,
                                bf16* __restrict__ qO,
                                bf16* __restrict__ kO,
                                bf16* __restrict__ vTO) {
  const int mt = blockIdx.x;          // 32-row tile index
  const int h  = blockIdx.y;
  const int z  = blockIdx.z;

  const bf16* wT = (z == 0) ? wqT : (z == 1) ? wkT : wvT;
  wT += (size_t)h * DH_ * D_;
  const float* bias = ((z == 0) ? bq : (z == 1) ? bk : bv) + h * DH_;

  const bf16* arow0 = xb + (size_t)(mt * 32 + 0)  * D_;
  const bf16* arow1 = xb + (size_t)(mt * 32 + 16) * D_;

  v8f acc[2][4] = {};
  v16bf aC[2], bC[4], aN[2], bN[4];

  aC[0] = load_frag(arow0, D_);
  aC[1] = load_frag(arow1, D_);
#pragma unroll
  for (int nt = 0; nt < 4; ++nt) bC[nt] = load_frag(wT + (size_t)nt * 16 * D_, D_);

  for (int k0 = 0; k0 < D_; k0 += 32) {
    const int kn = (k0 + 32 < D_) ? (k0 + 32) : 0;   // clamp: last prefetch harmless
    // issue next-step loads before this step's WMMAs
    aN[0] = load_frag(arow0 + kn, D_);
    aN[1] = load_frag(arow1 + kn, D_);
#pragma unroll
    for (int nt = 0; nt < 4; ++nt) bN[nt] = load_frag(wT + (size_t)nt * 16 * D_ + kn, D_);
#pragma unroll
    for (int nt = 0; nt < 4; ++nt) {
      acc[0][nt] = wmma_bf16(aC[0], bC[nt], acc[0][nt]);
      acc[1][nt] = wmma_bf16(aC[1], bC[nt], acc[1][nt]);
    }
    aC[0] = aN[0]; aC[1] = aN[1];
#pragma unroll
    for (int nt = 0; nt < 4; ++nt) bC[nt] = bN[nt];
  }

  const int l = lane_id();
  const int c = l & 15;
  const int half = l >> 4;
  if (z < 2) {
    bf16* dst = ((z == 0) ? qO : kO) + (size_t)h * S_ * DH_;
#pragma unroll
    for (int ms = 0; ms < 2; ++ms) {
#pragma unroll
      for (int nt = 0; nt < 4; ++nt) {
        const int col = nt * 16 + c;
        const float bb = bias[col];
#pragma unroll
        for (int r = 0; r < 8; ++r) {
          const int row = mt * 32 + ms * 16 + r + 8 * half;
          dst[(size_t)row * DH_ + col] = (bf16)(acc[ms][nt][r] + bb);
        }
      }
    }
  } else {  // store V transposed: [H][DH][S]
    bf16* dst = vTO + (size_t)h * DH_ * S_;
#pragma unroll
    for (int ms = 0; ms < 2; ++ms) {
#pragma unroll
      for (int nt = 0; nt < 4; ++nt) {
        const int col = nt * 16 + c;
        const float bb = bias[col];
#pragma unroll
        for (int r = 0; r < 8; ++r) {
          const int row = mt * 32 + ms * 16 + r + 8 * half;
          dst[(size_t)col * S_ + row] = (bf16)(acc[ms][nt][r] + bb);
        }
      }
    }
  }
}

// ---------------- flash attention ----------------
// grid: (S/16, H), block: 32. One wave owns 16 query rows, streams 4096 keys
// in 32-wide blocks with online softmax; ctx accumulated in fp32 registers.
// K fragments for block t0+32 are issued before the softmax of block t0
// (softmax VALU work hides the load latency); V fragments for block t0 are
// issued before the softmax and consumed after the LDS transpose.
__global__ void flash_attn_kernel(const bf16* __restrict__ q,
                                  const bf16* __restrict__ kmat,
                                  const bf16* __restrict__ vT,
                                  bf16* __restrict__ ctxb) {
  __shared__ __align__(16) bf16 lds_p[16 * 32];

  const int mt = blockIdx.x;
  const int h  = blockIdx.y;
  const bf16* qh  = q    + (size_t)h * S_ * DH_;
  const bf16* khp = kmat + (size_t)h * S_ * DH_;
  const bf16* vh  = vT   + (size_t)h * DH_ * S_;

  const int l = lane_id();
  const int c = l & 15;
  const int half = l >> 4;

  // Q fragments for this row tile (K-dim = DH = 64 -> two 32-wide A frags)
  const v16bf aq0 = load_frag(qh + (size_t)mt * 16 * DH_ + 0,  DH_);
  const v16bf aq1 = load_frag(qh + (size_t)mt * 16 * DH_ + 32, DH_);

  v8f acc[4] = {};
  float Mr[8], Lr[8];
#pragma unroll
  for (int r = 0; r < 8; ++r) { Mr[r] = -INFINITY; Lr[r] = 0.0f; }

  // preload K fragments for the first key block
  // kf[nt*2+j]: key rows t0+nt*16..+15, head-dim chunk j (e=0..31 / 32..63)
  v16bf kf[4], kn[4];
#pragma unroll
  for (int nt = 0; nt < 2; ++nt) {
    kf[nt * 2 + 0] = load_frag(khp + (size_t)(nt * 16) * DH_ + 0,  DH_);
    kf[nt * 2 + 1] = load_frag(khp + (size_t)(nt * 16) * DH_ + 32, DH_);
  }

  for (int t0 = 0; t0 < S_; t0 += 32) {
    // scores: 16 x 32 block (two 16x16 C frags), K-dim = 64
    v8f sc[2] = {};
    sc[0] = wmma_bf16(aq0, kf[0], sc[0]);
    sc[0] = wmma_bf16(aq1, kf[1], sc[0]);
    sc[1] = wmma_bf16(aq0, kf[2], sc[1]);
    sc[1] = wmma_bf16(aq1, kf[3], sc[1]);

    // issue current-block V loads and next-block K loads now; the softmax
    // below covers their latency.
    v16bf vf[4];
#pragma unroll
    for (int et = 0; et < 4; ++et)
      vf[et] = load_frag(vh + (size_t)(et * 16) * S_ + t0, S_);

    const int tn = (t0 + 32 < S_) ? (t0 + 32) : 0;  // clamp: harmless extra load
#pragma unroll
    for (int nt = 0; nt < 2; ++nt) {
      kn[nt * 2 + 0] = load_frag(khp + (size_t)(tn + nt * 16) * DH_ + 0,  DH_);
      kn[nt * 2 + 1] = load_frag(khp + (size_t)(tn + nt * 16) * DH_ + 32, DH_);
    }

    // online softmax per row r (rows r+8*half; columns live across 16 lanes)
#pragma unroll
    for (int r = 0; r < 8; ++r) {
      float s0 = sc[0][r] * 0.125f;   // 1/sqrt(DH)
      float s1 = sc[1][r] * 0.125f;
      float mx = fmaxf(s0, s1);
#pragma unroll
      for (int off = 8; off >= 1; off >>= 1)
        mx = fmaxf(mx, __shfl_xor(mx, off, 32));
      float newM = fmaxf(Mr[r], mx);
      float corr = __expf(Mr[r] - newM);
      float p0 = __expf(s0 - newM);
      float p1 = __expf(s1 - newM);
      float rs = p0 + p1;
#pragma unroll
      for (int off = 8; off >= 1; off >>= 1)
        rs += __shfl_xor(rs, off, 32);
      Lr[r] = Lr[r] * corr + rs;
      Mr[r] = newM;
#pragma unroll
      for (int et = 0; et < 4; ++et) acc[et][r] = acc[et][r] * corr;

      const int row = r + 8 * half;
      lds_p[row * 32 + c]      = (bf16)p0;
      lds_p[row * 32 + c + 16] = (bf16)p1;
    }

    // P (16x32) from LDS in A-fragment layout, then P @ V-block
    v16bf ap = load_frag(lds_p, 32);
#pragma unroll
    for (int et = 0; et < 4; ++et)
      acc[et] = wmma_bf16(ap, vf[et], acc[et]);

#pragma unroll
    for (int i = 0; i < 4; ++i) kf[i] = kn[i];
  }

  // epilogue: ctx = acc / L, stored to [S][H*DH] bf16
#pragma unroll
  for (int et = 0; et < 4; ++et) {
#pragma unroll
    for (int r = 0; r < 8; ++r) {
      const int row = mt * 16 + r + 8 * half;
      const int col = h * DH_ + et * 16 + c;
      ctxb[(size_t)row * D_ + col] = (bf16)(acc[et][r] / Lr[r]);
    }
  }
}

// ---------------- output projection ----------------
// grid: (S/32, D/64), block: 32. out = ctx @ wo + bo, fp32 result.
// 32x64 tile per wave, double-buffered fragments.
__global__ void out_proj_kernel(const bf16* __restrict__ ctxb,
                                const bf16* __restrict__ woT,
                                const float* __restrict__ bo,
                                float* __restrict__ out) {
  const int mt = blockIdx.x;
  const int n0 = blockIdx.y * 64;

  const bf16* arow0 = ctxb + (size_t)(mt * 32 + 0)  * D_;
  const bf16* arow1 = ctxb + (size_t)(mt * 32 + 16) * D_;
  const bf16* wrow  = woT + (size_t)n0 * D_;

  v8f acc[2][4] = {};
  v16bf aC[2], bC[4], aN[2], bN[4];

  aC[0] = load_frag(arow0, D_);
  aC[1] = load_frag(arow1, D_);
#pragma unroll
  for (int nt = 0; nt < 4; ++nt) bC[nt] = load_frag(wrow + (size_t)nt * 16 * D_, D_);

  for (int k0 = 0; k0 < D_; k0 += 32) {
    const int kn = (k0 + 32 < D_) ? (k0 + 32) : 0;
    aN[0] = load_frag(arow0 + kn, D_);
    aN[1] = load_frag(arow1 + kn, D_);
#pragma unroll
    for (int nt = 0; nt < 4; ++nt) bN[nt] = load_frag(wrow + (size_t)nt * 16 * D_ + kn, D_);
#pragma unroll
    for (int nt = 0; nt < 4; ++nt) {
      acc[0][nt] = wmma_bf16(aC[0], bC[nt], acc[0][nt]);
      acc[1][nt] = wmma_bf16(aC[1], bC[nt], acc[1][nt]);
    }
    aC[0] = aN[0]; aC[1] = aN[1];
#pragma unroll
    for (int nt = 0; nt < 4; ++nt) bC[nt] = bN[nt];
  }

  const int l = lane_id();
  const int c = l & 15;
  const int half = l >> 4;
#pragma unroll
  for (int ms = 0; ms < 2; ++ms) {
#pragma unroll
    for (int nt = 0; nt < 4; ++nt) {
      const int col = n0 + nt * 16 + c;
      const float bb = bo[col];
#pragma unroll
      for (int r = 0; r < 8; ++r) {
        const int row = mt * 32 + ms * 16 + r + 8 * half;
        out[(size_t)row * D_ + col] = acc[ms][nt][r] + bb;
      }
    }
  }
}

// ---------------- launch ----------------

extern "C" void kernel_launch(void* const* d_in, const int* in_sizes, int n_in,
                              void* d_out, int out_size, void* d_ws, size_t ws_size,
                              hipStream_t stream) {
  (void)in_sizes; (void)n_in; (void)out_size; (void)ws_size;
  const float* x  = (const float*)d_in[0];
  const float* wq = (const float*)d_in[1];
  const float* bq = (const float*)d_in[2];
  const float* wk = (const float*)d_in[3];
  const float* bk = (const float*)d_in[4];
  const float* wv = (const float*)d_in[5];
  const float* bv = (const float*)d_in[6];
  const float* wo = (const float*)d_in[7];
  const float* bo = (const float*)d_in[8];
  float* out = (float*)d_out;

  char* ws = (char*)d_ws;
  size_t off = 0;
  auto alloc = [&](size_t bytes) -> char* {
    char* p = ws + off;
    off += (bytes + 255) & ~(size_t)255;
    return p;
  };
  bf16* xb   = (bf16*)alloc((size_t)S_ * D_ * 2);          // x in bf16
  bf16* wqT  = (bf16*)alloc((size_t)H_ * DH_ * D_ * 2);    // [H][DH][D]
  bf16* wkT  = (bf16*)alloc((size_t)H_ * DH_ * D_ * 2);
  bf16* wvT  = (bf16*)alloc((size_t)H_ * DH_ * D_ * 2);
  bf16* woT  = (bf16*)alloc((size_t)D_ * D_ * 2);          // [N][K]
  bf16* qb   = (bf16*)alloc((size_t)H_ * S_ * DH_ * 2);    // [H][S][DH]
  bf16* kb   = (bf16*)alloc((size_t)H_ * S_ * DH_ * 2);    // [H][S][DH]
  bf16* vTb  = (bf16*)alloc((size_t)H_ * DH_ * S_ * 2);    // [H][DH][S]
  bf16* ctxb = (bf16*)alloc((size_t)S_ * D_ * 2);          // [S][D]

  cvt_f32_bf16<<<dim3(2048), 256, 0, stream>>>(x, xb, S_ * D_);
  cvt_wT<<<dim3(1152), 256, 0, stream>>>(wq, wqT);
  cvt_wT<<<dim3(1152), 256, 0, stream>>>(wk, wkT);
  cvt_wT<<<dim3(1152), 256, 0, stream>>>(wv, wvT);
  cvt_woT<<<dim3(1152), 256, 0, stream>>>(wo, woT);

  qkv_proj_kernel<<<dim3(S_ / 32, H_, 3), 32, 0, stream>>>(
      xb, wqT, wkT, wvT, bq, bk, bv, qb, kb, vTb);

  flash_attn_kernel<<<dim3(S_ / 16, H_), 32, 0, stream>>>(qb, kb, vTb, ctxb);

  out_proj_kernel<<<dim3(S_ / 32, D_ / 64), 32, 0, stream>>>(ctxb, woT, bo, out);
}
